// CGCNN_20306605376095
// MI455X (gfx1250) — compile-verified, matrix operand order checked by reading, not run on hardware
//
#include <hip/hip_runtime.h>

// ---------------- problem constants (match reference) ----------------
#define NN 50000
#define EE 1600000
#define FN 92
#define FE 41
#define CC 128
#define LL 3
#define GG 256
#define K2 (2 * CC + FE)          // 297 = concat(x_i, x_j, edge_attr)
#define KP 320                    // K padded to 10 chunks of 32 (bf16 WMMA K)
#define ZS 328                    // LDS row stride (halves), skewed for banks
#define NT ((EE + 15) / 16)       // 16-edge tiles

#if defined(__gfx1250__)
#define ASYNC_STAGE 1             // global_load_async_to_lds_b128 staging path
#endif

typedef __attribute__((ext_vector_type(16))) __bf16        v16bf;
typedef __attribute__((ext_vector_type(8)))  float         v8f;
typedef __attribute__((ext_vector_type(8)))  unsigned int  v8u;

// ---------------- helpers ----------------
__device__ __forceinline__ unsigned short f2bf(float f) {
    unsigned int u = __float_as_uint(f);
    u += 0x7fffu + ((u >> 16) & 1u);          // round-to-nearest-even
    return (unsigned short)(u >> 16);
}
__device__ __forceinline__ float softplus_f(float x) {
    return (x > 15.f) ? x : log1pf(__expf(x));
}
__device__ __forceinline__ float sigmoid_f(float x) {
    return 1.f / (1.f + __expf(-x));
}
__device__ __forceinline__ v16bf load_frag16(const unsigned short* p) {
    const unsigned int* q = (const unsigned int*)p;   // 16B-aligned by construction
    v8u r;
#pragma unroll
    for (int i = 0; i < 8; ++i) r[i] = q[i];
    return __builtin_bit_cast(v16bf, r);
}

// ---------------- weight conversion: fp32 [L][C][297] -> bf16 [L][C][320] ----------------
__global__ void k_convw(const float* __restrict__ Wf, const float* __restrict__ Ws,
                        unsigned short* __restrict__ wfb, unsigned short* __restrict__ wsb) {
    const int total = LL * CC * KP;
    for (int idx = blockIdx.x * blockDim.x + threadIdx.x; idx < total;
         idx += gridDim.x * blockDim.x) {
        const int k = idx % KP;
        const int r = idx / KP;                       // 0 .. L*C-1
        wfb[idx] = f2bf((k < K2) ? Wf[(size_t)r * K2 + k] : 0.f);
        wsb[idx] = f2bf((k < K2) ? Ws[(size_t)r * K2 + k] : 0.f);
    }
}

// ---------------- pre-FC: h = softplus(x @ Wpre^T + bpre); hb = bf16(h) ----------------
__global__ void k_pre(const float* __restrict__ x, const float* __restrict__ Wp,
                      const float* __restrict__ bp, float* __restrict__ h,
                      unsigned short* __restrict__ hb) {
    __shared__ float xr[FN];
    const int n = blockIdx.x, c = threadIdx.x;
    if (c < FN) xr[c] = x[(size_t)n * FN + c];
    __syncthreads();
    float acc = bp[c];
#pragma unroll 4
    for (int k = 0; k < FN; ++k) acc += xr[k] * Wp[(size_t)c * FN + k];
    const float r = softplus_f(acc);
    h[(size_t)n * CC + c]  = r;
    hb[(size_t)n * CC + c] = f2bf(r);
}

// ---------------- post-FC: hp = softplus(h @ Wpost^T + bpost) ----------------
__global__ void k_post(const float* __restrict__ h, const float* __restrict__ Wp,
                       const float* __restrict__ bp, float* __restrict__ hp) {
    __shared__ float xr[CC];
    const int n = blockIdx.x, c = threadIdx.x;
    xr[c] = h[(size_t)n * CC + c];
    __syncthreads();
    float acc = bp[c];
#pragma unroll 4
    for (int k = 0; k < CC; ++k) acc += xr[k] * Wp[(size_t)c * CC + k];
    hp[(size_t)n * CC + c] = softplus_f(acc);
}

// ---------------- zero fill ----------------
__global__ void k_zero(float* __restrict__ p, int n) {
    for (int i = blockIdx.x * blockDim.x + threadIdx.x; i < n;
         i += gridDim.x * blockDim.x) p[i] = 0.f;
}

// ---------------- residual + activation: h = softplus(h + agg); hb = bf16(h) ----------------
__global__ void k_resid(float* __restrict__ h, unsigned short* __restrict__ hb,
                        const float* __restrict__ agg, int n) {
    for (int i = blockIdx.x * blockDim.x + threadIdx.x; i < n;
         i += gridDim.x * blockDim.x) {
        const float r = softplus_f(h[i] + agg[i]);
        h[i]  = r;
        hb[i] = f2bf(r);
    }
}

// ---------------- CGConv edge kernel (bf16 WMMA + async LDS staging) ----------------
// Block = 256 threads = 8 waves. Wave w owns output cols [16w, 16w+16) and
// computes BOTH the Wf-gate and Ws-message 16x16 tiles for a 16-edge z-tile,
// fusing sigmoid*softplus and the atomic scatter to agg[dst].
// B (weight) fragments are register-resident across the grid-stride tile loop.
// Node-feature gathers go straight to LDS via GLOBAL_LOAD_ASYNC_TO_LDS_B128
// (ASYNCcnt), bypassing the VGPR return path; only the 41 edge_attr columns
// take the fp32->bf16 VALU conversion path.
__global__ __launch_bounds__(256)
void k_edge(const unsigned short* __restrict__ hb, const int* __restrict__ srcI,
            const int* __restrict__ dstI, const float* __restrict__ ea,
            const unsigned short* __restrict__ wfb,   // [C][KP] bf16, this layer
            const unsigned short* __restrict__ wsb,   // [C][KP] bf16, this layer
            const float* __restrict__ bfv_g, const float* __restrict__ bsv_g,
            float* __restrict__ agg) {
    extern __shared__ unsigned short zs[];            // [16][ZS] bf16 z-tile
    __shared__ int tdst[16];

    const int tid   = threadIdx.x;
    const int lane  = tid & 31;
    const int wave  = tid >> 5;
    const int n0    = wave * 16;                      // output col base
    const int colh  = lane & 15;                      // col within slice / A-row
    const int khalf = (lane >> 4) << 4;               // 0 or 16 (contiguous k = 16h + s order)

    // Preload this wave's B fragments for all 10 K-chunks, both matrices.
    v16bf wf_frag[10], ws_frag[10];
#pragma unroll
    for (int kc = 0; kc < 10; ++kc) {
        const int ks = kc * 32 + khalf;
        wf_frag[kc] = load_frag16(&wfb[(size_t)(n0 + colh) * KP + ks]);
        ws_frag[kc] = load_frag16(&wsb[(size_t)(n0 + colh) * KP + ks]);
    }
    const float bfv = bfv_g[n0 + colh];
    const float bsv = bsv_g[n0 + colh];

    const int row = tid >> 4;                         // staging row 0..15
    const int seg = tid & 15;                         // 16B segment 0..15

    for (int t = blockIdx.x; t < NT; t += gridDim.x) {
        const int eb = t * 16;
        // ---- stage z = [hb[dst] | hb[src] | bf16(edge_attr) | 0-pad] ----
        {
            const int e = eb + row;
            const int s = srcI[e];
            const int d = dstI[e];
            if (seg == 0) tdst[row] = d;
#ifdef ASYNC_STAGE
            // per-lane async copy: 8 bf16 (16B) per transfer, engine writes LDS
            asm volatile("global_load_async_to_lds_b128 %0, %1, off"
                         :: "v"((unsigned)(unsigned long long)&zs[row * ZS + seg * 8]),
                            "v"((unsigned long long)&hb[(size_t)d * CC + seg * 8])
                         : "memory");
            asm volatile("global_load_async_to_lds_b128 %0, %1, off"
                         :: "v"((unsigned)(unsigned long long)&zs[row * ZS + CC + seg * 8]),
                            "v"((unsigned long long)&hb[(size_t)s * CC + seg * 8])
                         : "memory");
#else
            *(uint4*)&zs[row * ZS + seg * 8] =
                *(const uint4*)&hb[(size_t)d * CC + seg * 8];
            *(uint4*)&zs[row * ZS + CC + seg * 8] =
                *(const uint4*)&hb[(size_t)s * CC + seg * 8];
#endif
            // edge_attr columns 256..296 (+ zero pad to 319), overlapped with async
            const int ca = 2 * CC + (tid & 63);
#pragma unroll
            for (int rr = 0; rr < 4; ++rr) {
                const int r2 = (tid >> 6) + rr * 4;
                const float v = (ca < K2)
                    ? ea[(size_t)(eb + r2) * FE + (ca - 2 * CC)] : 0.f;
                zs[r2 * ZS + ca] = f2bf(v);
            }
#ifdef ASYNC_STAGE
            asm volatile("s_wait_asynccnt 0x0" ::: "memory");
#endif
        }
        __syncthreads();

        // ---- dual bf16 WMMA over 10 K-chunks ----
        v8f accF = {0.f, 0.f, 0.f, 0.f, 0.f, 0.f, 0.f, 0.f};
        v8f accS = {0.f, 0.f, 0.f, 0.f, 0.f, 0.f, 0.f, 0.f};
#pragma unroll
        for (int kc = 0; kc < 10; ++kc) {
            const int ks = kc * 32 + khalf;
            v16bf a = load_frag16(&zs[colh * ZS + ks]);
            accF = __builtin_amdgcn_wmma_f32_16x16x32_bf16(
                false, a, false, wf_frag[kc], (short)0, accF, false, false);
            accS = __builtin_amdgcn_wmma_f32_16x16x32_bf16(
                false, a, false, ws_frag[kc], (short)0, accS, false, false);
        }

        // ---- fuse gates and scatter-add to destination nodes ----
#pragma unroll
        for (int r = 0; r < 8; ++r) {
            const int m = (lane < 16) ? r : (r + 8);  // edge row in tile
            const float mval = sigmoid_f(accF[r] + bfv) * softplus_f(accS[r] + bsv);
            atomicAdd(&agg[(size_t)tdst[m] * CC + n0 + colh], mval);
        }
        __syncthreads();                              // protect zs/tdst reuse
    }
}

// ---------------- global mean pool (segment sum) ----------------
__global__ void k_pool(const float* __restrict__ hp, const int* __restrict__ batch,
                       float* __restrict__ sums, float* __restrict__ cnts) {
    const int n = blockIdx.x, c = threadIdx.x;
    const int g = batch[n];
    atomicAdd(&sums[(size_t)g * CC + c], hp[(size_t)n * CC + c]);
    if (c == 0) atomicAdd(&cnts[g], 1.f);
}

// ---------------- head: out[g] = (pooled @ W_out^T) + b_out ----------------
__global__ void k_head(const float* __restrict__ sums, const float* __restrict__ cnts,
                       const float* __restrict__ Wo, const float* __restrict__ bo,
                       float* __restrict__ out) {
    __shared__ float red[CC];
    const int g = blockIdx.x, c = threadIdx.x;
    const float cnt = fmaxf(cnts[g], 1.f);
    red[c] = (sums[(size_t)g * CC + c] / cnt) * Wo[c];
    __syncthreads();
    for (int s = CC / 2; s > 0; s >>= 1) {
        if (c < s) red[c] += red[c + s];
        __syncthreads();
    }
    if (c == 0) out[g] = red[0] + bo[0];
}

// ---------------- launcher ----------------
extern "C" void kernel_launch(void* const* d_in, const int* in_sizes, int n_in,
                              void* d_out, int out_size, void* d_ws, size_t ws_size,
                              hipStream_t stream) {
    const float* x     = (const float*)d_in[0];
    const int*   ei    = (const int*)d_in[1];
    const float* ea    = (const float*)d_in[2];
    const int*   batch = (const int*)d_in[3];
    const float* Wpre  = (const float*)d_in[4];
    const float* bpre  = (const float*)d_in[5];
    const float* Wf    = (const float*)d_in[6];
    const float* bfb   = (const float*)d_in[7];
    const float* Ws    = (const float*)d_in[8];
    const float* bsb   = (const float*)d_in[9];
    const float* Wpost = (const float*)d_in[10];
    const float* bpost = (const float*)d_in[11];
    const float* Wout  = (const float*)d_in[12];
    const float* bout  = (const float*)d_in[13];
    float* out = (float*)d_out;

    const int* srcI = ei;          // edge_index[0]
    const int* dstI = ei + EE;     // edge_index[1]

    // workspace carve-up (~65 MB, all 16B aligned)
    char* ws = (char*)d_ws;
    float* h    = (float*)ws;  ws += sizeof(float) * (size_t)NN * CC;
    float* agg  = (float*)ws;  ws += sizeof(float) * (size_t)NN * CC;   // also post-FC out
    float* sums = (float*)ws;  ws += sizeof(float) * (size_t)GG * CC;
    float* cnts = (float*)ws;  ws += sizeof(float) * (size_t)GG;
    unsigned short* hb  = (unsigned short*)ws; ws += sizeof(unsigned short) * (size_t)NN * CC;
    unsigned short* wfb = (unsigned short*)ws; ws += sizeof(unsigned short) * (size_t)LL * CC * KP;
    unsigned short* wsb = (unsigned short*)ws;

    k_convw<<<512, 256, 0, stream>>>(Wf, Ws, wfb, wsb);
    k_pre<<<NN, CC, 0, stream>>>(x, Wpre, bpre, h, hb);

    const size_t zbytes = (size_t)16 * ZS * sizeof(unsigned short);
    for (int l = 0; l < LL; ++l) {
        k_zero<<<2048, 256, 0, stream>>>(agg, NN * CC);
        k_edge<<<4096, 256, zbytes, stream>>>(hb, srcI, dstI, ea,
                                              wfb + (size_t)l * CC * KP,
                                              wsb + (size_t)l * CC * KP,
                                              bfb + l * CC, bsb + l * CC, agg);
        k_resid<<<2048, 256, 0, stream>>>(h, hb, agg, NN * CC);
    }

    k_post<<<NN, CC, 0, stream>>>(h, Wpost, bpost, agg);   // agg reused as hp
    k_zero<<<64, 256, 0, stream>>>(sums, GG * CC);
    k_zero<<<1, 256, 0, stream>>>(cnts, GG);
    k_pool<<<NN, CC, 0, stream>>>(agg, batch, sums, cnts);
    k_head<<<GG, CC, 0, stream>>>(sums, cnts, Wout, bout, out);
}
